// CausalSelfAttention_19404662243825
// MI455X (gfx1250) — matrix operand order repaired and checked by reading, not compiled, not run on hardware
//
#include <hip/hip_runtime.h>
#include <hip/hip_bf16.h>
#include <math.h>

typedef __bf16 bf16_t;
typedef __attribute__((ext_vector_type(16))) __bf16 v16bf;
typedef __attribute__((ext_vector_type(8)))  __bf16 v8bf;
typedef __attribute__((ext_vector_type(8)))  float  v8f;

#define B_  4
#define T_  2048
#define C_  1024
#define H_  16
#define HS_ 64
#define M_  (B_ * T_)     /* 8192 */
#define N1_ (3 * C_)      /* 3072 */

// ---------------- WMMA helpers ----------------

__device__ __forceinline__ v8f wmma_bf16(v16bf a, v16bf b, v8f c) {
  return __builtin_amdgcn_wmma_f32_16x16x32_bf16(false, a, false, b, (short)0, c,
                                                 false, false);
}

// A-operand (16x32 bf16) load from row-major [*, ldk] matrix.
// lane holds row m; K = h*16 + e (contiguous 16).
__device__ __forceinline__ v16bf load_a_tile(const bf16_t* __restrict__ A, int ldk,
                                             int m, int k0, int lane) {
  int h = lane >> 4;
  const bf16_t* p = A + (size_t)m * ldk + k0 + h * 16;
  v8bf lo = *(const v8bf*)p;
  v8bf hi = *(const v8bf*)(p + 8);
  v16bf a;
#pragma unroll
  for (int i = 0; i < 8; ++i) { a[i] = lo[i]; a[i + 8] = hi[i]; }
  return a;
}

// B-operand (32x16 bf16) load from "transposed" matrix where row n holds K
// contiguously: lane holds col n, K = (lane/16)*16 + e.
__device__ __forceinline__ v16bf load_b_tile(const bf16_t* __restrict__ Bt, int ldk,
                                             int n, int k0, int lane) {
  int h = lane >> 4;
  return *(const v16bf*)(Bt + (size_t)n * ldk + k0 + h * 16);
}

// ---------------- conversion kernels ----------------

__global__ void cvt_f32_bf16(const float* __restrict__ in, bf16_t* __restrict__ out,
                             int n) {
  int i = blockIdx.x * blockDim.x + threadIdx.x;
  if (i < n) out[i] = (bf16_t)in[i];
}

// src: [K, N] f32 row-major  ->  dst: [N, K] bf16 row-major
__global__ void transpose_f32_bf16(const float* __restrict__ src,
                                   bf16_t* __restrict__ dst, int K, int N) {
  int i = blockIdx.x * blockDim.x + threadIdx.x;
  if (i >= N * K) return;
  int n = i / K, k = i % K;
  dst[i] = (bf16_t)src[(size_t)k * N + n];
}

// ---------------- GEMM1: qkv = x @ w_kqv, 32x64 strip per wave ----------------

__global__ __launch_bounds__(256) void gemm_qkv(const bf16_t* __restrict__ xb,
                                                const bf16_t* __restrict__ wT,
                                                bf16_t* __restrict__ qb,
                                                bf16_t* __restrict__ kb,
                                                bf16_t* __restrict__ vtb) {
  int lane = threadIdx.x & 31;
  int wave = threadIdx.x >> 5;
  int lm = lane & 15, h = lane >> 4;
  int m0 = blockIdx.x * 32;
  int n0 = blockIdx.y * 512 + wave * 64;

  v8f acc[2][4] = {};
  for (int k0 = 0; k0 < C_; k0 += 32) {
    if (k0 + 32 < C_) {
      __builtin_prefetch(xb + (size_t)(m0 + lm) * C_ + k0 + 32, 0, 1);
      __builtin_prefetch(wT + (size_t)(n0 + lm) * C_ + k0 + 32, 0, 1);
    }
    v16bf a0 = load_a_tile(xb, C_, m0 + lm, k0, lane);
    v16bf a1 = load_a_tile(xb, C_, m0 + 16 + lm, k0, lane);
#pragma unroll
    for (int t = 0; t < 4; ++t) {
      v16bf b = load_b_tile(wT, C_, n0 + t * 16 + lm, k0, lane);
      acc[0][t] = wmma_bf16(a0, b, acc[0][t]);
      acc[1][t] = wmma_bf16(a1, b, acc[1][t]);
    }
  }
#pragma unroll
  for (int rt = 0; rt < 2; ++rt) {
#pragma unroll
    for (int t = 0; t < 4; ++t) {
      int n = n0 + t * 16 + lm;
      int sel = n / C_, cc = n % C_;
      int hh = cc >> 6, d = cc & 63;
#pragma unroll
      for (int r = 0; r < 8; ++r) {
        int mrow = m0 + rt * 16 + r + h * 8;
        int b = mrow >> 11, tt = mrow & (T_ - 1);
        bf16_t val = (bf16_t)acc[rt][t][r];
        if (sel == 0)
          qb[(((size_t)(b * H_ + hh) * T_ + tt) << 6) + d] = val;       // [B,H,T,HS]
        else if (sel == 1)
          kb[(((size_t)(b * H_ + hh) * T_ + tt) << 6) + d] = val;       // [B,H,T,HS]
        else
          vtb[((size_t)(b * H_ + hh) * HS_ + d) * T_ + tt] = val;       // [B,H,HS,T]
      }
    }
  }
}

// ------------- flash attention: one wave per (b,h,32 q-rows) -------------

__global__ __launch_bounds__(32) void attn_kernel(const bf16_t* __restrict__ q,
                                                  const bf16_t* __restrict__ k,
                                                  const bf16_t* __restrict__ vt,
                                                  bf16_t* __restrict__ ctx) {
  __shared__ bf16_t lp[32][32];  // P staging (D-layout f32 -> A-layout bf16)
  int lane = threadIdx.x & 31;
  int lm = lane & 15, h = lane >> 4;
  int bh = blockIdx.x;               // b*H + h
  int b = bh >> 4, hh = bh & 15;
  int q0 = blockIdx.y * 32;

  const bf16_t* qh  = q  + (size_t)bh * T_ * HS_;
  const bf16_t* kh  = k  + (size_t)bh * T_ * HS_;
  const bf16_t* vth = vt + (size_t)bh * HS_ * T_;

  v16bf qa[2][2];  // [row-tile][k-chunk]
#pragma unroll
  for (int rt = 0; rt < 2; ++rt)
#pragma unroll
    for (int c = 0; c < 2; ++c)
      qa[rt][c] = load_a_tile(qh, HS_, q0 + rt * 16 + lm, c * 32, lane);

  float mrow[2][8], lrow[2][8];
#pragma unroll
  for (int rt = 0; rt < 2; ++rt)
#pragma unroll
    for (int r = 0; r < 8; ++r) { mrow[rt][r] = -INFINITY; lrow[rt][r] = 0.f; }
  v8f Oacc[2][4] = {};

  int kend = q0 + 32;  // exclusive causal bound
  for (int kb = 0; kb < kend; kb += 32) {
    // K B-tiles, shared by both q row-tiles
    v16bf bkt[2][2];
#pragma unroll
    for (int s = 0; s < 2; ++s)
#pragma unroll
      for (int c = 0; c < 2; ++c)
        bkt[s][c] = load_b_tile(kh, HS_, kb + s * 16 + lm, c * 32, lane);

#pragma unroll
    for (int rt = 0; rt < 2; ++rt) {
      float sv[2][8];
#pragma unroll
      for (int s = 0; s < 2; ++s) {
        v8f sacc = {};
#pragma unroll
        for (int c = 0; c < 2; ++c) sacc = wmma_bf16(qa[rt][c], bkt[s][c], sacc);
        int col = kb + s * 16 + lm;
#pragma unroll
        for (int r = 0; r < 8; ++r) {
          int row = q0 + rt * 16 + r + h * 8;
          float v = sacc[r] * 8.0f;  // reference multiplies by sqrt(HS)
          sv[s][r] = (col <= row) ? v : -INFINITY;
        }
      }
      // online softmax row stats (row values live across lanes [h*16, h*16+15])
#pragma unroll
      for (int r = 0; r < 8; ++r) {
        float t = fmaxf(sv[0][r], sv[1][r]);
#pragma unroll
        for (int xm = 1; xm < 16; xm <<= 1) t = fmaxf(t, __shfl_xor(t, xm, 32));
        float nm = fmaxf(mrow[rt][r], t);
        float alpha = __expf(mrow[rt][r] - nm);
        mrow[rt][r] = nm;
        float p0 = __expf(sv[0][r] - nm);
        float p1 = __expf(sv[1][r] - nm);
        lp[rt * 16 + r + h * 8][lm]      = (bf16_t)p0;
        lp[rt * 16 + r + h * 8][16 + lm] = (bf16_t)p1;
        float ps = p0 + p1;
#pragma unroll
        for (int xm = 1; xm < 16; xm <<= 1) ps += __shfl_xor(ps, xm, 32);
        lrow[rt][r] = lrow[rt][r] * alpha + ps;
#pragma unroll
        for (int t2 = 0; t2 < 4; ++t2) Oacc[rt][t2][r] *= alpha;
      }
    }
    __syncthreads();  // single-wave workgroup: near-free, orders LDS
    // P (16x32 per row-tile) as A-operands from LDS
    v16bf pa[2];
#pragma unroll
    for (int rt = 0; rt < 2; ++rt) {
      v8bf plo = *(const v8bf*)&lp[rt * 16 + lm][h * 16];
      v8bf phi = *(const v8bf*)&lp[rt * 16 + lm][h * 16 + 8];
#pragma unroll
      for (int i = 0; i < 8; ++i) { pa[rt][i] = plo[i]; pa[rt][i + 8] = phi[i]; }
    }
    // O += P @ V ; V B-tiles shared by both row-tiles (V^T stored [HS,T])
#pragma unroll
    for (int t = 0; t < 4; ++t) {
      v16bf bv = *(const v16bf*)(vth + (size_t)(t * 16 + lm) * T_ + kb + h * 16);
      Oacc[0][t] = wmma_bf16(pa[0], bv, Oacc[0][t]);
      Oacc[1][t] = wmma_bf16(pa[1], bv, Oacc[1][t]);
    }
    __syncthreads();
  }
  // normalize and write context in [B,T,C] bf16
#pragma unroll
  for (int rt = 0; rt < 2; ++rt)
#pragma unroll
    for (int t = 0; t < 4; ++t)
#pragma unroll
      for (int r = 0; r < 8; ++r) {
        int row = q0 + rt * 16 + r + h * 8;
        float val = Oacc[rt][t][r] / lrow[rt][r];
        ctx[(size_t)(b * T_ + row) * C_ + hh * 64 + t * 16 + lm] = (bf16_t)val;
      }
}

// ---------------- GEMM2: out = ctx @ w_proj + b_proj, 32x64 per wave ----------------

__global__ __launch_bounds__(256) void gemm_proj(const bf16_t* __restrict__ ctx,
                                                 const bf16_t* __restrict__ wT,
                                                 const float* __restrict__ bias,
                                                 float* __restrict__ out) {
  int lane = threadIdx.x & 31;
  int wave = threadIdx.x >> 5;
  int lm = lane & 15, h = lane >> 4;
  int m0 = blockIdx.x * 32;
  int n0 = blockIdx.y * 512 + wave * 64;

  v8f acc[2][4] = {};
  for (int k0 = 0; k0 < C_; k0 += 32) {
    if (k0 + 32 < C_) {
      __builtin_prefetch(ctx + (size_t)(m0 + lm) * C_ + k0 + 32, 0, 1);
      __builtin_prefetch(wT + (size_t)(n0 + lm) * C_ + k0 + 32, 0, 1);
    }
    v16bf a0 = load_a_tile(ctx, C_, m0 + lm, k0, lane);
    v16bf a1 = load_a_tile(ctx, C_, m0 + 16 + lm, k0, lane);
#pragma unroll
    for (int t = 0; t < 4; ++t) {
      v16bf b = load_b_tile(wT, C_, n0 + t * 16 + lm, k0, lane);
      acc[0][t] = wmma_bf16(a0, b, acc[0][t]);
      acc[1][t] = wmma_bf16(a1, b, acc[1][t]);
    }
  }
#pragma unroll
  for (int rt = 0; rt < 2; ++rt)
#pragma unroll
    for (int t = 0; t < 4; ++t) {
      int n = n0 + t * 16 + lm;
      float bn = bias[n];
#pragma unroll
      for (int r = 0; r < 8; ++r) {
        int mrow = m0 + rt * 16 + r + h * 8;
        out[(size_t)mrow * C_ + n] = acc[rt][t][r] + bn;
      }
    }
}

// ---------------- host launcher ----------------

extern "C" void kernel_launch(void* const* d_in, const int* in_sizes, int n_in,
                              void* d_out, int out_size, void* d_ws, size_t ws_size,
                              hipStream_t stream) {
  (void)in_sizes; (void)n_in; (void)out_size; (void)ws_size;
  const float* x      = (const float*)d_in[0];
  const float* w_kqv  = (const float*)d_in[1];
  const float* w_proj = (const float*)d_in[2];
  const float* b_proj = (const float*)d_in[3];
  float* out = (float*)d_out;

  char* ws = (char*)d_ws;
  size_t off = 0;
  bf16_t* xb  = (bf16_t*)(ws + off); off += (size_t)M_ * C_ * 2;    // 16 MB
  bf16_t* wkT = (bf16_t*)(ws + off); off += (size_t)N1_ * C_ * 2;   //  6 MB
  bf16_t* wpT = (bf16_t*)(ws + off); off += (size_t)C_ * C_ * 2;    //  2 MB
  bf16_t* qb  = (bf16_t*)(ws + off); off += (size_t)M_ * C_ * 2;    // 16 MB
  bf16_t* kb  = (bf16_t*)(ws + off); off += (size_t)M_ * C_ * 2;    // 16 MB
  bf16_t* vtb = (bf16_t*)(ws + off); off += (size_t)M_ * C_ * 2;    // 16 MB
  bf16_t* ctx = (bf16_t*)(ws + off); off += (size_t)M_ * C_ * 2;    // 16 MB

  cvt_f32_bf16<<<(M_ * C_) / 256, 256, 0, stream>>>(x, xb, M_ * C_);
  transpose_f32_bf16<<<(N1_ * C_) / 256, 256, 0, stream>>>(w_kqv, wkT, C_, N1_);
  transpose_f32_bf16<<<(C_ * C_) / 256, 256, 0, stream>>>(w_proj, wpT, C_, C_);

  gemm_qkv<<<dim3(M_ / 32, N1_ / 512), 256, 0, stream>>>(xb, wkT, qb, kb, vtb);
  attn_kernel<<<dim3(B_ * H_, T_ / 32), 32, 0, stream>>>(qb, kb, vtb, ctx);
  gemm_proj<<<dim3(M_ / 32, C_ / 512), 256, 0, stream>>>(ctx, wpT, b_proj, out);
}
